// FourierKAN_INR_37048387895773
// MI455X (gfx1250) — compile-verified, matrix operand order checked by reading, not run on hardware
//
#include <hip/hip_runtime.h>
#include <hip/hip_bf16.h>

// ---------------------------------------------------------------------------
// Fused FourierKAN + SIREN-like INR forward for MI455X (gfx1250, wave32).
// GEMMs: v_wmma_f32_16x16x32_bf16. Weight panels staged by the Tensor Data
// Mover (tensor_load_to_lds, TENSORcnt), double-buffered against compute.
// ---------------------------------------------------------------------------

#define NPTS    131072
#define IN_F    3
#define HID     256
#define GRIDK   64
#define OMEGA   30.0f
#define LN_EPS  1e-5f

#define M_TILE  64          // rows per workgroup
#define STRIDE  520         // activation LDS row stride (bf16), rows 16B-aligned
#define NPANEL  32          // output columns per staged weight panel

typedef __attribute__((ext_vector_type(16))) __bf16 bf16x16;
typedef __attribute__((ext_vector_type(8)))  __bf16 bf16x8;
typedef __attribute__((ext_vector_type(8)))  float  f32x8;

// bf16 weight arena layout in d_ws (element offsets)
#define OFF_KAN 0            // 256 x 384  (cos|sin features, row-major)
#define OFF_W1  98304        // 512 x 256
#define OFF_W2  229376       // 512 x 512
#define OFF_W3  491520       // 512 x 512
#define OFF_W4  753664       // 1024 x 512
#define WTS_TOTAL 1277952

// Padded weight-panel row stride: 8 bf16 pad every 128 bf16 (TDM pad:
// interval = 64 DWORDs = 256B, amount = 4 DWORDs = 16B).
#define KPAD(K)   ((K) + (((K) >> 7) << 3))
#define KADJ(k)   ((k) + (((k) >> 7) << 3))

// LDS byte layout (within dynamic segment)
#define BUF_BYTES   (M_TILE * STRIDE * 2)           // 66560
#define WB_ELEMS    (NPANEL * KPAD(512))            // 17408 bf16 (max K)
#define WB_BYTES    (WB_ELEMS * 2)                  // 34816
#define OFFB_BUF0   0
#define OFFB_BUF1   (BUF_BYTES)                     // 66560
#define OFFB_WB0    (2 * BUF_BYTES)                 // 133120
#define OFFB_WB1    (2 * BUF_BYTES + WB_BYTES)      // 167936
#define OFFB_F32    (2 * BUF_BYTES + 2 * WB_BYTES)  // 202752
#define SMEM_BYTES  (OFFB_F32 + (3072 + 512 + 192 + 192) * 4)  // 218624

// ---------------------------------------------------------------------------
// TDM availability
// ---------------------------------------------------------------------------
#if defined(__has_builtin)
#if __has_builtin(__builtin_amdgcn_tensor_load_to_lds) && \
    __has_builtin(__builtin_amdgcn_s_wait_tensorcnt)
#define USE_TDM 1
#endif
#endif
#ifndef USE_TDM
#define USE_TDM 0
#endif

#if USE_TDM
typedef __attribute__((ext_vector_type(4))) unsigned int u32x4;
typedef __attribute__((ext_vector_type(8))) int          i32x8;
typedef __attribute__((ext_vector_type(4))) int          i32x4;

// Issue one TDM load of a 32 x K bf16 panel (row-major, contiguous in global)
// into LDS at byte offset lds_off, with 16B pad inserted every 256B so rows
// land with stride KPAD(K) elements. 2D tensor => groups 2/3 unused (zero).
__device__ __forceinline__ void tdm_load_panel(const __bf16* gsrc,
                                               unsigned lds_off, int K) {
    unsigned long long ga = (unsigned long long)(size_t)gsrc;
    u32x4 g0;
    g0.x = 1u;                                   // count=1, user descriptor
    g0.y = lds_off;                              // lds_addr (bytes)
    g0.z = (unsigned)ga;                         // global_addr[31:0]
    g0.w = (unsigned)((ga >> 32) & 0x01ffffffu)  // global_addr[56:32]
         | (2u << 30);                           // type=2 ("image")
    unsigned td0  = (unsigned)K;                 // tensor_dim0 (elements)
    unsigned td1  = 32u;                         // tensor_dim1 (rows)
    unsigned st0  = (unsigned)K;                 // tensor_dim0_stride
    i32x8 g1;
    g1[0] = (int)((1u << 16)     // data_size = 1 -> 2 bytes
                | (1u << 20)     // pad_enable
                | (5u << 22)     // pad_interval = 64 DWORDs (256B)
                | (3u << 25));   // pad_amount   = 4 DWORDs (16B)
    g1[1] = (int)((td0 & 0xffffu) << 16);                        // dim0 lo16
    g1[2] = (int)(((td0 >> 16) & 0xffffu) | ((td1 & 0xffffu) << 16));
    g1[3] = (int)(((td1 >> 16) & 0xffffu) | ((td0 & 0xffffu) << 16)); // tile_dim0=K
    g1[4] = (int)(td1 & 0xffffu);                // tile_dim1 = 32, tile_dim2 = 0
    g1[5] = (int)st0;                            // dim0_stride[31:0]
    g1[6] = (int)((st0 & 0xffffu) << 16);        // dim0_stride hi=0 | dim1_stride lo16
    g1[7] = 0;
    i32x4 gz = {0, 0, 0, 0};
#if __clang_major__ >= 23
    i32x8 gz8 = {0, 0, 0, 0, 0, 0, 0, 0};
    __builtin_amdgcn_tensor_load_to_lds(g0, g1, gz, gz, gz8, 0);
#else
    __builtin_amdgcn_tensor_load_to_lds(g0, g1, gz, gz, 0);
#endif
}
#endif // USE_TDM

// ---------------------------------------------------------------------------
// Weight prep: fp32 -> bf16; KAN coeffs folded to (256 x 384) row-major where
// column f: s=f/192 (0=cos,1=sin), i=(f%192)/64, g=f%64  <->  fc[s][o][i][g].
// ---------------------------------------------------------------------------
__global__ __launch_bounds__(256) void prep_weights(
    const float* __restrict__ fc, const float* __restrict__ w1,
    const float* __restrict__ w2, const float* __restrict__ w3,
    const float* __restrict__ w4, __bf16* __restrict__ wts)
{
    int idx = blockIdx.x * 256 + threadIdx.x;
    if (idx >= WTS_TOTAL) return;
    float v;
    if (idx < OFF_W1) {
        int o = idx / 384, f = idx % 384;
        int s = f / 192, rem = f % 192, i = rem >> 6, g = rem & 63;
        v = fc[(((s * HID + o) * IN_F) + i) * GRIDK + g];
    } else if (idx < OFF_W2) {
        v = w1[idx - OFF_W1];
    } else if (idx < OFF_W3) {
        v = w2[idx - OFF_W2];
    } else if (idx < OFF_W4) {
        v = w3[idx - OFF_W3];
    } else {
        v = w4[idx - OFF_W4];
    }
    wts[idx] = (__bf16)v;
}

// ---------------------------------------------------------------------------
// Device helpers
// ---------------------------------------------------------------------------
__device__ __forceinline__ float act_sine(float z) {
    // (z + tanh(30 z)) * sigmoid(z)
    float t = tanhf(OMEGA * z);
    float s = __builtin_amdgcn_rcpf(1.0f + __expf(-z));
    return (z + t) * s;
}

// One 16x32 bf16 operand fragment from LDS.
// ISA 16-bit A layout: lane<16 -> K {kc..kc+7, kc+16..kc+23},
// lane>=16 -> K {kc+8..kc+15, kc+24..kc+31}; kc = caller-adjusted column base.
__device__ __forceinline__ bf16x16 load_frag(const __bf16* base, int row,
                                             int stride, int kc, int lane) {
    const __bf16* p = base + row * stride + kc + ((lane >> 4) << 3);
    bf16x8 lo = *(const bf16x8*)p;
    bf16x8 hi = *(const bf16x8*)(p + 16);
    return __builtin_shufflevector(lo, hi, 0, 1, 2, 3, 4, 5, 6, 7,
                                   8, 9, 10, 11, 12, 13, 14, 15);
}

// ---------------------------------------------------------------------------
// Core GEMM: Out[M_TILE x O] = epi( Ain[M_TILE x K] * Wg^T + bias ).
// Wg: row-major (O x K) bf16 in global (L2-hot). Weight panels (32 x K)
// double-buffered in LDS via TDM. FUSED_OUT: apply act + contract with
// wout (3 x 1024, f32 LDS) into outacc instead of storing activations.
// ---------------------------------------------------------------------------
template <bool FUSED_OUT>
__device__ __forceinline__ void gemm_core(
    const __bf16* __restrict__ Ain, const __bf16* __restrict__ Wg,
    const float* __restrict__ bias, __bf16* __restrict__ Out,
    const float* woutb, float* outacc,
    int K, int O, __bf16* wb0, __bf16* wb1,
    unsigned wb0_off, unsigned wb1_off,
    int tid, int lane, int wave, bool actflag)
{
    const int KP = KPAD(K);
    const int waveM = wave >> 1, waveN = wave & 1;
    const int m0 = waveM * 16, nl0 = waveN * 16;
    const int hi = lane >> 4, ln = lane & 15;
    const int arow = m0 + ln;   // A fragment row
    const int brow = nl0 + ln;  // B fragment row (= local output column)
    const int panels = O / NPANEL;
    const int total = NPANEL * K;

    auto compute = [&](const __bf16* wb, int p) {
        const int colg = p * NPANEL + nl0 + ln;
        float bval = bias[colg];
        f32x8 acc;
#pragma unroll
        for (int r = 0; r < 8; ++r) acc[r] = bval;

        for (int k0 = 0; k0 < K; k0 += 32) {
            bf16x16 a = load_frag(Ain, arow, STRIDE, k0, lane);
            bf16x16 b = load_frag(wb, brow, KP, KADJ(k0), lane);
            acc = __builtin_amdgcn_wmma_f32_16x16x32_bf16(
                false, a, false, b, (short)0, acc, false, false);
        }

        if (FUSED_OUT) {
#pragma unroll
            for (int r = 0; r < 8; ++r) {
                float h = act_sine(acc[r]);
                int row = m0 + r + 8 * hi;
#pragma unroll
                for (int o = 0; o < 3; ++o) {
                    float v = h * woutb[o * 1024 + colg];
                    v += __shfl_xor(v, 1, 32);
                    v += __shfl_xor(v, 2, 32);
                    v += __shfl_xor(v, 4, 32);
                    v += __shfl_xor(v, 8, 32);
                    if (ln == 0) atomicAdd(&outacc[row * 3 + o], v);
                }
            }
        } else {
#pragma unroll
            for (int r = 0; r < 8; ++r) {
                float v = acc[r];
                if (actflag) v = act_sine(v);
                Out[(m0 + r + 8 * hi) * STRIDE + colg] = (__bf16)v;
            }
        }
    };

#if USE_TDM
    // ---- async TDM pipeline: DMA panel p+1 overlaps compute of panel p ----
    if (wave == 0) tdm_load_panel(Wg, wb0_off, K);
    for (int p = 0; p < panels; ++p) {
        const __bf16* wb = (p & 1) ? wb1 : wb0;
        if (wave == 0) __builtin_amdgcn_s_wait_tensorcnt(0);
        __syncthreads();                       // panel p visible to all waves
        if (wave == 0 && p + 1 < panels)
            tdm_load_panel(Wg + (size_t)(p + 1) * total,
                           ((p + 1) & 1) ? wb1_off : wb0_off, K);
        compute(wb, p);
        __syncthreads();                       // done reading before overwrite
    }
#else
    // ---- fallback: synchronous single-buffer staging (same padded layout) --
    (void)wb1; (void)wb0_off; (void)wb1_off;
    for (int p = 0; p < panels; ++p) {
        const __bf16* wsrc = Wg + (size_t)p * total;
        if (p + 1 < panels) __builtin_prefetch(Wg + (size_t)(p + 1) * total, 0, 0);
        for (int i = tid * 8; i < total; i += 256 * 8) {
            int r = i / K, c = i % K;
            *(uint4*)&wb0[r * KP + KADJ(c)] = *(const uint4*)&wsrc[i];
        }
        __syncthreads();
        compute(wb0, p);
        __syncthreads();
    }
#endif
}

// ---------------------------------------------------------------------------
// Main fused kernel: one 64-point tile per workgroup, 8 waves (4M x 2N tiles).
// ---------------------------------------------------------------------------
__global__ __launch_bounds__(256) void fkan_inr_kernel(
    const float* __restrict__ coords, const __bf16* __restrict__ wts,
    const float* __restrict__ fkan_bias,
    const float* __restrict__ ln_g, const float* __restrict__ ln_b,
    const float* __restrict__ b1, const float* __restrict__ b2,
    const float* __restrict__ b3, const float* __restrict__ b4,
    const float* __restrict__ wout, const float* __restrict__ bout,
    float* __restrict__ out)
{
    extern __shared__ __align__(16) unsigned char smem_raw[];
    __bf16* buf0   = (__bf16*)(smem_raw + OFFB_BUF0);   // 64 x 520 bf16
    __bf16* buf1   = (__bf16*)(smem_raw + OFFB_BUF1);   // 64 x 520 bf16
    __bf16* wb0    = (__bf16*)(smem_raw + OFFB_WB0);    // 32 x KPAD(512) bf16
    __bf16* wb1    = (__bf16*)(smem_raw + OFFB_WB1);    // 32 x KPAD(512) bf16
    float*  woutb  = (float*)(smem_raw + OFFB_F32);     // 3 x 1024 f32
    float*  lnpar  = woutb + 3072;                      // gamma[256] beta[256]
    float*  outacc = lnpar + 512;                       // 64 x 3 f32
    float*  cbuf   = outacc + 192;                      // 64 x 3 f32 coords

    const unsigned lds_base = __builtin_amdgcn_groupstaticsize();
    const unsigned wb0_off = lds_base + OFFB_WB0;
    const unsigned wb1_off = lds_base + OFFB_WB1;

    const int tid  = threadIdx.x;
    const int lane = tid & 31;
    const int wave = tid >> 5;
    const size_t base = (size_t)blockIdx.x * M_TILE;

    // ---- stage coords, wout, LN params, output-bias accumulators ----
    if (tid < M_TILE * IN_F) cbuf[tid] = coords[base * IN_F + tid];
    for (int i = tid; i < 3072; i += 256) woutb[i] = wout[i];
    if (tid < 256) { lnpar[tid] = ln_g[tid]; lnpar[256 + tid] = ln_b[tid]; }
    if (tid < M_TILE * 3) outacc[tid] = bout[tid % 3];
    __syncthreads();

    // ---- Fourier features: cos/sin(k * x_i), 64 rows x 384 cols, bf16 ----
    for (int idx = tid; idx < M_TILE * 384; idx += 256) {
        int m = idx / 384, f = idx % 384;
        int s = f / 192, rem = f % 192, i = rem >> 6, g = rem & 63;
        float ang = cbuf[m * IN_F + i] * (float)(g + 1);
        float v = s ? __sinf(ang) : __cosf(ang);
        buf0[m * STRIDE + f] = (__bf16)v;
    }
    __syncthreads();

    // ---- KAN GEMM (K=384 -> 256), bias = fkan_bias, no activation ----
    gemm_core<false>(buf0, wts + OFF_KAN, fkan_bias, buf1, nullptr, nullptr,
                     384, 256, wb0, wb1, wb0_off, wb1_off, tid, lane, wave, false);

    // ---- LayerNorm over 256 columns, in-place on buf1 ----
    if (tid < M_TILE) {
        __bf16* rowp = buf1 + tid * STRIDE;
        float mu = 0.f;
        for (int c = 0; c < HID; ++c) mu += (float)rowp[c];
        mu *= (1.0f / HID);
        float var = 0.f;
        for (int c = 0; c < HID; ++c) {
            float d = (float)rowp[c] - mu;
            var += d * d;
        }
        var *= (1.0f / HID);
        float rs = __builtin_amdgcn_rsqf(var + LN_EPS);
        for (int c = 0; c < HID; ++c) {
            float d = ((float)rowp[c] - mu) * rs * lnpar[c] + lnpar[256 + c];
            rowp[c] = (__bf16)d;
        }
    }
    __syncthreads();

    // ---- sine layers 1..3 (ping-pong buf1 <-> buf0) ----
    gemm_core<false>(buf1, wts + OFF_W1, b1, buf0, nullptr, nullptr,
                     256, 512, wb0, wb1, wb0_off, wb1_off, tid, lane, wave, true);
    gemm_core<false>(buf0, wts + OFF_W2, b2, buf1, nullptr, nullptr,
                     512, 512, wb0, wb1, wb0_off, wb1_off, tid, lane, wave, true);
    gemm_core<false>(buf1, wts + OFF_W3, b3, buf0, nullptr, nullptr,
                     512, 512, wb0, wb1, wb0_off, wb1_off, tid, lane, wave, true);

    // ---- layer 4 (512 -> 1024) fused with output projection (1024 -> 3) ----
    gemm_core<true>(buf0, wts + OFF_W4, b4, nullptr, woutb, outacc,
                    512, 1024, wb0, wb1, wb0_off, wb1_off, tid, lane, wave, true);
    __syncthreads();

    if (tid < M_TILE * 3) out[base * 3 + tid] = outacc[tid];
}

// ---------------------------------------------------------------------------
// Host entry
// ---------------------------------------------------------------------------
extern "C" void kernel_launch(void* const* d_in, const int* in_sizes, int n_in,
                              void* d_out, int out_size, void* d_ws, size_t ws_size,
                              hipStream_t stream) {
    const float* coords    = (const float*)d_in[0];
    const float* fc        = (const float*)d_in[1];
    const float* fkan_bias = (const float*)d_in[2];
    const float* ln_g      = (const float*)d_in[3];
    const float* ln_b      = (const float*)d_in[4];
    const float* w1        = (const float*)d_in[5];
    const float* b1        = (const float*)d_in[6];
    const float* w2        = (const float*)d_in[7];
    const float* b2        = (const float*)d_in[8];
    const float* w3        = (const float*)d_in[9];
    const float* b3        = (const float*)d_in[10];
    const float* w4        = (const float*)d_in[11];
    const float* b4        = (const float*)d_in[12];
    const float* wout      = (const float*)d_in[13];
    const float* bout      = (const float*)d_in[14];
    __bf16* wts = (__bf16*)d_ws;

    prep_weights<<<(WTS_TOTAL + 255) / 256, 256, 0, stream>>>(fc, w1, w2, w3, w4, wts);

    (void)hipFuncSetAttribute((const void*)fkan_inr_kernel,
                              hipFuncAttributeMaxDynamicSharedMemorySize,
                              SMEM_BYTES);
    fkan_inr_kernel<<<NPTS / M_TILE, 256, SMEM_BYTES, stream>>>(
        coords, wts, fkan_bias, ln_g, ln_b, b1, b2, b3, b4, wout, bout,
        (float*)d_out);
}
